// MultiHeadAttention_44109314130338
// MI455X (gfx1250) — matrix operand order repaired and checked by reading, not compiled, not run on hardware
//
#include <hip/hip_runtime.h>

// ---------------------------------------------------------------------------
// MI455X (gfx1250) multi-head "joint d,t" attention.
//   Stage 1: proj_kernel  x3   : Y = W @ X + b  -> bf16, layout [B][H][P][D*T]
//   Stage 2: energy_kernel     : E[p][q] = sum_f K[p][f]*Q[q][f]  (fp32)
//                                LDS staging via Tensor Data Mover (TDM) with
//                                hardware padding reproducing the [128][40] layout
//   Stage 3: softmax_kernel    : attn = softmax(E/SCALE) rows; bf16 in-place
//   Stage 4: out_kernel        : Out[q][f] = sum_p attn[p][q]*V[p][f] -> (B,C,P,T)
// All GEMMs: v_wmma_f32_16x16x32_bf16, 128x128 block tile, 8 waves of 64x32.
// Workspace: 3 * 128MiB (K/Q/V bf16) + 128MiB (energy fp32 / attn bf16) = 512MiB.
// ---------------------------------------------------------------------------

typedef __attribute__((ext_vector_type(16))) __bf16 v16bf;
typedef __attribute__((ext_vector_type(8)))  float  v8f;
typedef __attribute__((ext_vector_type(4)))  unsigned int v4u;
typedef __attribute__((ext_vector_type(8)))  int   v8i;
typedef __attribute__((ext_vector_type(4)))  int   v4i;

#define NB 4
#define NH 8
#define NP 1024
#define NT 64
#define ND 32
#define NC 256
#define NF 2048                 // D*T, the joint contraction length
#define NPT 65536               // P*T
#define INV_SCALE 0.02209708691207961f   // 1/sqrt(D*T)

#define BM 128
#define BN 128
#define BK 32
#define LDK 40                  // padded LDS k-stride (80B rows, 16B aligned)

#if __has_builtin(__builtin_amdgcn_tensor_load_to_lds)
#define USE_TDM 1
#else
#define USE_TDM 0
#endif

union FragU { v16bf v; uint4 q[2]; };
union BF4U  { uint2 u; __bf16 b[4]; };

static __device__ __forceinline__ __bf16 f2bf(float f) {
    union { float f; unsigned u; } x; x.f = f;
    unsigned short s = (unsigned short)((x.u + 0x7FFFu + ((x.u >> 16) & 1u)) >> 16);
    return __builtin_bit_cast(__bf16, s);
}

// Packed fp32 -> 2 x bf16 store (4B aligned dst). Uses the HW packed convert
// when clang exposes it for gfx1250; software round-to-nearest-even otherwise.
static __device__ __forceinline__ void store2bf(__bf16* dst, float a, float b) {
#if __has_builtin(__builtin_amdgcn_cvt_pk_bf16_f32)
    auto r = __builtin_amdgcn_cvt_pk_bf16_f32(a, b);
    static_assert(sizeof(r) == 4, "cvt_pk_bf16_f32 must produce 32 bits");
    *reinterpret_cast<decltype(r)*>(dst) = r;
#else
    dst[0] = f2bf(a); dst[1] = f2bf(b);
#endif
}

static __device__ __forceinline__ v8f vzero8() {
    v8f z;
    #pragma unroll
    for (int i = 0; i < 8; ++i) z[i] = 0.0f;
    return z;
}

// One 32-deep K-step of the 128x128 tile for this wave: 4x2 WMMA frags.
static __device__ __forceinline__ void mma_step(
    const __bf16 (*As)[LDK], const __bf16 (*Bts)[LDK],
    v8f acc[4][2], int wm, int wn, int lane)
{
    const int fr    = lane & 15;
    const int koffA = (lane >> 4) * 8;    // A lanes 0-15: K 0..7 & 16..23 ; 16-31: K 8..15 & 24..31
    const int khB   = (lane >> 4) * 8;    // B lanes 0-15: K 0..7 & 16..23 ; 16-31: K 8..15 & 24..31
    FragU a[4], b[2];
    #pragma unroll
    for (int i = 0; i < 4; ++i) {
        const __bf16* row = As[wm + i * 16 + fr];
        a[i].q[0] = *(const uint4*)(row + koffA);
        a[i].q[1] = *(const uint4*)(row + 16 + koffA);
    }
    #pragma unroll
    for (int j = 0; j < 2; ++j) {
        const __bf16* row = Bts[wn + j * 16 + fr];
        b[j].q[0] = *(const uint4*)(row + khB);
        b[j].q[1] = *(const uint4*)(row + 16 + khB);
    }
    #pragma unroll
    for (int i = 0; i < 4; ++i)
        #pragma unroll
        for (int j = 0; j < 2; ++j)
            acc[i][j] = __builtin_amdgcn_wmma_f32_16x16x32_bf16(
                false, a[i].v, false, b[j].v, (short)0, acc[i][j], false, false);
}

#if USE_TDM
// Issue one TDM 2D tile load: 128 rows x 32 bf16 (row stride = NF elements)
// into LDS with HW padding of 4 dwords per 16 dwords -> [128][LDK] layout.
static __device__ __forceinline__ void tdm_tile_load(
    unsigned lds_byte_addr, unsigned long long gaddr)
{
    v4u g0;
    g0[0] = 1u;                                       // count=1, is_restore=0
    g0[1] = lds_byte_addr;                            // lds_addr (bytes)
    g0[2] = (unsigned)gaddr;                          // global_addr[31:0]
    g0[3] = (unsigned)((gaddr >> 32) & 0x1FFFFFFull)  // global_addr[56:32]
          | 0x80000000u;                              // type=2 ("image")
    v8i g1;
    g1[0] = (int)((1u << 16)      // data_size = 1 -> 2 bytes
                | (1u << 20)      // pad_enable
                | (3u << 22)      // pad_interval: 2^(3+1)=16 dwords (one 32-elem row)
                | (3u << 25));    // pad_amount: 3+1 = 4 dwords (8 bf16)
    g1[1] = (int)((unsigned)NF << 16);   // tensor_dim0 = 2048 (bits 79:48, low half)
    g1[2] = (int)((unsigned)NP << 16);   // tensor_dim0 hi = 0; tensor_dim1 lo = 1024
    g1[3] = (int)(32u << 16);            // tensor_dim1 hi = 0; tile_dim0 = 32
    g1[4] = 128;                          // tile_dim1 = 128; tile_dim2 = 0
    g1[5] = NF;                           // tensor_dim0_stride = 2048 (low 32)
    g1[6] = 0;                            // stride0 hi; tensor_dim1_stride lo (unused, 2D)
    g1[7] = 0;
    v4i z4; z4[0] = z4[1] = z4[2] = z4[3] = 0;        // groups 2/3 unused (2D tensor)
#if defined(__clang_major__) && __clang_major__ >= 23
    v8i z8;
    #pragma unroll
    for (int i = 0; i < 8; ++i) z8[i] = 0;
    __builtin_amdgcn_tensor_load_to_lds(g0, g1, z4, z4, z8, 0);
#else
    __builtin_amdgcn_tensor_load_to_lds(g0, g1, z4, z4, 0);
#endif
}
#endif

// ---------------------------------------------------------------------------
// Stage 1: projection. A = W (C x C fp32, row-major over c), B = x (C x P*T fp32).
// Output bf16 at [((b*H+h)*P+p)*NF + d*T + t], o = h*32+d, n = p*64+t.
// ---------------------------------------------------------------------------
__global__ __launch_bounds__(256) void mha_proj_kernel(
    const float* __restrict__ W, const float* __restrict__ bias,
    const float* __restrict__ x, __bf16* __restrict__ out)
{
    __shared__ __align__(16) __bf16 As[BM][LDK];
    __shared__ __align__(16) __bf16 Bts[BN][LDK];

    const int tid  = threadIdx.x;
    const int lane = tid & 31;
    const int wave = tid >> 5;
    const int wm = (wave & 1) * 64;
    const int wn = (wave >> 1) * 32;
    const int bb = blockIdx.z;
    const int m0 = blockIdx.y * BM;
    const int n0 = blockIdx.x * BN;
    const float* Xb = x + (size_t)bb * NC * NPT;

    v8f acc[4][2];
    #pragma unroll
    for (int i = 0; i < 4; ++i)
        #pragma unroll
        for (int j = 0; j < 2; ++j) acc[i][j] = vzero8();

    for (int k0 = 0; k0 < NC; k0 += BK) {
        // A tile: 128 x 32, fp32 -> bf16, direct (packed converts)
        #pragma unroll
        for (int it = 0; it < 4; ++it) {
            int idx = it * 1024 + tid * 4;
            int r = idx >> 5, c = idx & 31;
            float4 f = *(const float4*)(W + (size_t)(m0 + r) * NC + k0 + c);
            store2bf(&As[r][c],     f.x, f.y);
            store2bf(&As[r][c + 2], f.z, f.w);
        }
        // B tile: 32 x 128, fp32 -> bf16, transposed into Bts[n][k]
        #pragma unroll
        for (int it = 0; it < 4; ++it) {
            int idx = it * 1024 + tid * 4;
            int kr = idx >> 7, nc = idx & 127;
            float4 f = *(const float4*)(Xb + (size_t)(k0 + kr) * NPT + n0 + nc);
            Bts[nc + 0][kr] = f2bf(f.x); Bts[nc + 1][kr] = f2bf(f.y);
            Bts[nc + 2][kr] = f2bf(f.z); Bts[nc + 3][kr] = f2bf(f.w);
        }
        if (k0 + BK < NC) {
            __builtin_prefetch(Xb + (size_t)(k0 + BK + (tid >> 3)) * NPT + n0 + (tid & 7) * 16, 0, 1);
        }
        __syncthreads();
        mma_step(As, Bts, acc, wm, wn, lane);
        __syncthreads();
    }

    const int rowoff = (lane >> 4) * 8;
    #pragma unroll
    for (int i = 0; i < 4; ++i) {
        #pragma unroll
        for (int j = 0; j < 2; ++j) {
            int gn = n0 + wn + j * 16 + (lane & 15);
            int p = gn >> 6, t = gn & 63;
            #pragma unroll
            for (int e = 0; e < 8; ++e) {
                int gm = m0 + wm + i * 16 + rowoff + e;     // output channel o
                float v = acc[i][j][e] + bias[gm];
                int h = gm >> 5, d = gm & 31;
                out[(((size_t)bb * NH + h) * NP + p) * NF + d * NT + t] = f2bf(v);
            }
        }
    }
}

// ---------------------------------------------------------------------------
// Stage 2: energy[p][q] = sum_f K[p][f] * Q[q][f].  Both bf16 row-major (ld=NF).
// LDS tiles filled by the Tensor Data Mover (wave 0 issues, all waves barrier).
// Output fp32 energy[bh][p][q].
// ---------------------------------------------------------------------------
__global__ __launch_bounds__(256) void mha_energy_kernel(
    const __bf16* __restrict__ Kb, const __bf16* __restrict__ Qb,
    float* __restrict__ eng)
{
    __shared__ __align__(16) __bf16 As[BM][LDK];
    __shared__ __align__(16) __bf16 Bts[BN][LDK];

    const int tid  = threadIdx.x;
    const int lane = tid & 31;
    const int wave = tid >> 5;
    const int wm = (wave & 1) * 64;
    const int wn = (wave >> 1) * 32;
    const int bh = blockIdx.z;
    const int m0 = blockIdx.y * BM;     // p
    const int n0 = blockIdx.x * BN;     // q
    const __bf16* Ab = Kb + (size_t)bh * NP * NF;
    const __bf16* Bb = Qb + (size_t)bh * NP * NF;

    v8f acc[4][2];
    #pragma unroll
    for (int i = 0; i < 4; ++i)
        #pragma unroll
        for (int j = 0; j < 2; ++j) acc[i][j] = vzero8();

#if USE_TDM
    const unsigned ldsA = (unsigned)(uintptr_t)(&As[0][0]);
    const unsigned ldsB = (unsigned)(uintptr_t)(&Bts[0][0]);
    const unsigned long long gA0 =
        (unsigned long long)(uintptr_t)Ab + (unsigned long long)m0 * NF * 2ull;
    const unsigned long long gB0 =
        (unsigned long long)(uintptr_t)Bb + (unsigned long long)n0 * NF * 2ull;
#endif

    for (int k0 = 0; k0 < NF; k0 += BK) {
#if USE_TDM
        if (wave == 0) {
            tdm_tile_load(ldsA, gA0 + (unsigned long long)k0 * 2ull);
            tdm_tile_load(ldsB, gB0 + (unsigned long long)k0 * 2ull);
            __builtin_amdgcn_s_wait_tensorcnt(0);
        }
#else
        #pragma unroll
        for (int it = 0; it < 4; ++it) {
            int idx = it * 1024 + tid * 4;
            int r = idx >> 5, c = idx & 31;
            *(uint2*)&As[r][c]  = *(const uint2*)(Ab + (size_t)(m0 + r) * NF + k0 + c);
            *(uint2*)&Bts[r][c] = *(const uint2*)(Bb + (size_t)(n0 + r) * NF + k0 + c);
        }
#endif
        __syncthreads();
        mma_step(As, Bts, acc, wm, wn, lane);
        __syncthreads();
    }

    const int rowoff = (lane >> 4) * 8;
    float* Eb = eng + (size_t)bh * NP * NP;
    #pragma unroll
    for (int i = 0; i < 4; ++i)
        #pragma unroll
        for (int j = 0; j < 2; ++j) {
            int gn = n0 + wn + j * 16 + (lane & 15);
            #pragma unroll
            for (int e = 0; e < 8; ++e) {
                int gm = m0 + wm + i * 16 + rowoff + e;
                Eb[(size_t)gm * NP + gn] = acc[i][j][e];
            }
        }
}

// ---------------------------------------------------------------------------
// Stage 3: row softmax over q. Reads fp32 row (4KB), writes bf16 attn row
// in-place at the start of the same 4KB region (row stride 2048 bf16).
// ---------------------------------------------------------------------------
__global__ __launch_bounds__(256) void mha_softmax_kernel(float* __restrict__ eng)
{
    __shared__ float red[8];
    const int tid  = threadIdx.x;
    const int lane = tid & 31;
    const int wid  = tid >> 5;
    float* base = eng + (size_t)blockIdx.x * NP;

    float4 vv = *(const float4*)(base + tid * 4);
    vv.x *= INV_SCALE; vv.y *= INV_SCALE; vv.z *= INV_SCALE; vv.w *= INV_SCALE;

    float m = fmaxf(fmaxf(vv.x, vv.y), fmaxf(vv.z, vv.w));
    #pragma unroll
    for (int off = 16; off > 0; off >>= 1) m = fmaxf(m, __shfl_xor(m, off, 32));
    if (lane == 0) red[wid] = m;
    __syncthreads();
    float rm = red[0];
    #pragma unroll
    for (int i = 1; i < 8; ++i) rm = fmaxf(rm, red[i]);
    __syncthreads();

    float p0 = __expf(vv.x - rm), p1 = __expf(vv.y - rm);
    float p2 = __expf(vv.z - rm), p3 = __expf(vv.w - rm);
    float s = (p0 + p1) + (p2 + p3);
    #pragma unroll
    for (int off = 16; off > 0; off >>= 1) s += __shfl_xor(s, off, 32);
    if (lane == 0) red[wid] = s;
    __syncthreads();
    float rs = 0.0f;
    #pragma unroll
    for (int i = 0; i < 8; ++i) rs += red[i];
    float inv = 1.0f / rs;

    __bf16* arow = (__bf16*)base;   // bf16 row occupies first half of the fp32 row
    store2bf(&arow[tid * 4],     p0 * inv, p1 * inv);
    store2bf(&arow[tid * 4 + 2], p2 * inv, p3 * inv);
}

// ---------------------------------------------------------------------------
// Stage 4: Out[q][f] = sum_p attn[p][q] * V[p][f].
// A = attn^T (staged with LDS transpose; attn row stride = NF bf16 in the
// reused energy buffer), B = V (staged transposed). Output fp32 (B,C,P,T).
// ---------------------------------------------------------------------------
__global__ __launch_bounds__(256) void mha_out_kernel(
    const __bf16* __restrict__ attn, const __bf16* __restrict__ Vb,
    float* __restrict__ out)
{
    __shared__ __align__(16) __bf16 As[BM][LDK];
    __shared__ __align__(16) __bf16 Bts[BN][LDK];

    const int tid  = threadIdx.x;
    const int lane = tid & 31;
    const int wave = tid >> 5;
    const int wm = (wave & 1) * 64;
    const int wn = (wave >> 1) * 32;
    const int bh = blockIdx.z;
    const int m0 = blockIdx.y * BM;     // q
    const int n0 = blockIdx.x * BN;     // f = d*64+t
    const __bf16* At = attn + (size_t)bh * NP * NF;   // [p][q], row stride NF
    const __bf16* Vt = Vb   + (size_t)bh * NP * NF;   // [p][f], row stride NF

    v8f acc[4][2];
    #pragma unroll
    for (int i = 0; i < 4; ++i)
        #pragma unroll
        for (int j = 0; j < 2; ++j) acc[i][j] = vzero8();

    for (int k0 = 0; k0 < NP; k0 += BK) {   // k = p
        #pragma unroll
        for (int it = 0; it < 4; ++it) {
            int idx = it * 1024 + tid * 4;
            int kr = idx >> 7, nc = idx & 127;
            BF4U a; a.u = *(const uint2*)(At + (size_t)(k0 + kr) * NF + m0 + nc);
            As[nc + 0][kr] = a.b[0]; As[nc + 1][kr] = a.b[1];
            As[nc + 2][kr] = a.b[2]; As[nc + 3][kr] = a.b[3];
            BF4U v; v.u = *(const uint2*)(Vt + (size_t)(k0 + kr) * NF + n0 + nc);
            Bts[nc + 0][kr] = v.b[0]; Bts[nc + 1][kr] = v.b[1];
            Bts[nc + 2][kr] = v.b[2]; Bts[nc + 3][kr] = v.b[3];
        }
        if (k0 + BK < NP) {
            __builtin_prefetch(At + (size_t)(k0 + BK + (tid & 31)) * NF + m0, 0, 1);
            __builtin_prefetch(Vt + (size_t)(k0 + BK + (tid & 31)) * NF + n0, 0, 1);
        }
        __syncthreads();
        mma_step(As, Bts, acc, wm, wn, lane);
        __syncthreads();
    }

    const int rowoff = (lane >> 4) * 8;
    const int bb = bh >> 3, hh = bh & 7;
    #pragma unroll
    for (int i = 0; i < 4; ++i)
        #pragma unroll
        for (int j = 0; j < 2; ++j) {
            int gn = n0 + wn + j * 16 + (lane & 15);
            int d = gn >> 6, t = gn & 63;
            #pragma unroll
            for (int e = 0; e < 8; ++e) {
                int gq = m0 + wm + i * 16 + rowoff + e;
                out[(((size_t)bb * NC + hh * ND + d) * NP + gq) * NT + t] = acc[i][j][e];
            }
        }
}

// ---------------------------------------------------------------------------
extern "C" void kernel_launch(void* const* d_in, const int* in_sizes, int n_in,
                              void* d_out, int out_size, void* d_ws, size_t ws_size,
                              hipStream_t stream) {
    const float* x  = (const float*)d_in[0];
    const float* Wk = (const float*)d_in[1];
    const float* bk = (const float*)d_in[2];
    const float* Wq = (const float*)d_in[3];
    const float* bq = (const float*)d_in[4];
    const float* Wv = (const float*)d_in[5];
    const float* bv = (const float*)d_in[6];
    float* out = (float*)d_out;

    const size_t nKQV = (size_t)NB * NH * NP * NF;   // 67,108,864 elements
    __bf16* Kb = (__bf16*)d_ws;
    __bf16* Qb = Kb + nKQV;
    __bf16* Vb = Qb + nKQV;
    float*  eng = (float*)(Vb + nKQV);               // 128MiB, reused for bf16 attn
    // total ws: 512 MiB

    dim3 blk(256);
    dim3 gProj(NPT / BN, NC / BM, NB);               // 512 x 2 x 4
    mha_proj_kernel<<<gProj, blk, 0, stream>>>(Wk, bk, x, Kb);
    mha_proj_kernel<<<gProj, blk, 0, stream>>>(Wq, bq, x, Qb);
    mha_proj_kernel<<<gProj, blk, 0, stream>>>(Wv, bv, x, Vb);
    mha_energy_kernel<<<dim3(NP / BN, NP / BM, NB * NH), blk, 0, stream>>>(Kb, Qb, eng);
    mha_softmax_kernel<<<dim3(NB * NH * NP), blk, 0, stream>>>(eng);
    mha_out_kernel<<<dim3(NF / BN, NP / BM, NB * NH), blk, 0, stream>>>(
        (const __bf16*)eng, Vb, out);
}
